// BidirectionalMambaModel_5042291605978
// MI455X (gfx1250) — matrix operand order repaired and checked by reading, not compiled, not run on hardware
//
#include <hip/hip_runtime.h>
#include <hip/hip_bf16.h>
#include <math.h>

#define D_MODEL 768
#define D_INNER 1536
#define D_STATE 16
#define D_CONV  4
#define DT_RANK 48
#define N_LAYER 24
#define BATCH   2
#define SEQ     1024
#define NTOK    (BATCH * SEQ)            // 2048
#define XPROJ_N (DT_RANK + 2 * D_STATE)  // 80

typedef __attribute__((ext_vector_type(16))) __bf16 v16bf;
typedef __attribute__((ext_vector_type(8)))  float  v8f;
typedef __attribute__((ext_vector_type(4))) unsigned int uint32x4;
typedef __attribute__((ext_vector_type(8))) int int32x8;
typedef __attribute__((ext_vector_type(4))) int int32x4;

__device__ __forceinline__ float silu_(float x) { return x / (1.f + __expf(-x)); }
__device__ __forceinline__ float softplus_(float x) {
  return (x > 20.f) ? x : log1pf(__expf(x));
}

__device__ __forceinline__ v8f wmma_bf16_(v16bf a, v16bf b, v8f c) {
  // (neg_a, A, neg_b, B, c_mod, C, reuse_a, reuse_b)
  return __builtin_amdgcn_wmma_f32_16x16x32_bf16(false, a, false, b, (short)0, c,
                                                 false, false);
}

// ---------------------------------------------------------------------------
// Tensor Data Mover: 2D tile load Global -> LDS (D# per ISA 08_async_tensor §8)
// rows x rowlen f32 tile; LDS receives rows contiguously (rowlen*4B each).
// ---------------------------------------------------------------------------
__device__ __forceinline__ void tdm_load_2d_(unsigned lds_off,
                                             const float* __restrict__ gaddr,
                                             int rows, int rowlen, int stride) {
  unsigned long long ga = (unsigned long long)(uintptr_t)gaddr;
  uint32x4 g0;
  g0[0] = 1u;                                   // count=1 (valid), user mode
  g0[1] = lds_off;                              // lds_addr
  g0[2] = (unsigned)(ga & 0xFFFFFFFFu);         // global_addr[31:0]
  g0[3] = (unsigned)((ga >> 32) & 0x1FFFFFFu)   // global_addr[56:32]
          | (2u << 30);                         // type=2 ("image")
  int32x8 g1;
  g1[0] = (int)(2u << 16);                      // data_size=2 (4 bytes)
  g1[1] = (int)(((unsigned)rowlen & 0xFFFFu) << 16);        // tensor_dim0[15:0]
  g1[2] = (int)((((unsigned)rowlen >> 16) & 0xFFFFu)        // tensor_dim0[31:16]
          | (((unsigned)rows & 0xFFFFu) << 16));            // tensor_dim1[15:0]
  g1[3] = (int)((((unsigned)rows >> 16) & 0xFFFFu)          // tensor_dim1[31:16]
          | (((unsigned)rowlen & 0xFFFFu) << 16));          // tile_dim0
  g1[4] = (int)((unsigned)rows & 0xFFFFu);                  // tile_dim1 (dim2=0)
  g1[5] = (int)(unsigned)stride;                            // dim0_stride[31:0]
  g1[6] = 0;                                                // stride msbs
  g1[7] = 0;
  int32x4 z4 = {0, 0, 0, 0};
#if defined(__clang_major__) && (__clang_major__ >= 23)
  int32x8 z8 = {0, 0, 0, 0, 0, 0, 0, 0};
  __builtin_amdgcn_tensor_load_to_lds(g0, g1, z4, z4, z8, 0);
#else
  __builtin_amdgcn_tensor_load_to_lds(g0, g1, z4, z4, 0);
#endif
}

// A fragment: 16x32 bf16 tile (MxK). ISA layout: lanes 0-15 hold M=0..15 with
// K pairs {0..7,16..23}; lanes 16-31 hold same M with K pairs {8..15,24..31}.
__device__ __forceinline__ v16bf frag_a_(const float* __restrict__ A, int lda,
                                         int tileM, int k0, int K, int lane,
                                         bool guard) {
  const float* p = A + (size_t)(tileM + (lane & 15)) * lda;
  const int kh = (lane >> 4) << 3;  // 0 or 8
  v16bf a;
#pragma unroll
  for (int j = 0; j < 4; ++j) {
    int ka = k0 + kh + 2 * j;
    int kb = ka + 16;
    a[2 * j]         = (__bf16)((!guard || ka     < K) ? p[ka]     : 0.f);
    a[2 * j + 1]     = (__bf16)((!guard || ka + 1 < K) ? p[ka + 1] : 0.f);
    a[8 + 2 * j]     = (__bf16)((!guard || kb     < K) ? p[kb]     : 0.f);
    a[8 + 2 * j + 1] = (__bf16)((!guard || kb + 1 < K) ? p[kb + 1] : 0.f);
  }
  return a;
}

// B fragment: 32x16 bf16 tile (KxN) gathered from W[N,K] row-major (C = A*W^T).
__device__ __forceinline__ v16bf frag_b_(const float* __restrict__ W, int ldw,
                                         int tileN, int k0, int K, int lane,
                                         bool guard) {
  const float* p = W + (size_t)(tileN + (lane & 15)) * ldw;
  const int kh = (lane >> 4) << 3;  // 0 or 8
  v16bf b;
#pragma unroll
  for (int j = 0; j < 4; ++j) {
    int k = k0 + kh + 2 * j;
    int kb = k + 16;
    b[2 * j]         = (__bf16)((!guard || k      < K) ? p[k]      : 0.f);
    b[2 * j + 1]     = (__bf16)((!guard || k + 1  < K) ? p[k + 1]  : 0.f);
    b[8 + 2 * j]     = (__bf16)((!guard || kb     < K) ? p[kb]     : 0.f);
    b[8 + 2 * j + 1] = (__bf16)((!guard || kb + 1 < K) ? p[kb + 1] : 0.f);
  }
  return b;
}

// EPI: 0 = plain store, 1 = softplus(c + bias[n]), 2 = dual store (C and C2)
template <int EPI>
__device__ __forceinline__ void store_tile_(v8f acc, float* __restrict__ C, int ldc,
                                            float* __restrict__ C2, int ldc2,
                                            const float* __restrict__ bias,
                                            int tileM, int tileN, int lane) {
  int n  = tileN + (lane & 15);
  int m0 = tileM + ((lane >> 4) << 3);  // C/D layout: lane half selects M+8
  float bn = (EPI == 1) ? bias[n] : 0.f;
#pragma unroll
  for (int r = 0; r < 8; ++r) {
    float v = acc[r];
    if (EPI == 1) v = softplus_(v + bn);
    C[(size_t)(m0 + r) * ldc + n] = v;
    if (EPI == 2) C2[(size_t)(m0 + r) * ldc2 + n] = v;
  }
}

// ---------------------------------------------------------------------------
// Direct-load WMMA GEMM (handles K tails / ragged N): one wave, 32x32 tile.
// ---------------------------------------------------------------------------
template <int EPI>
__global__ void __launch_bounds__(32)
gemm_wmma_nt(const float* __restrict__ A, int lda,
             const float* __restrict__ W, int ldw,
             const float* __restrict__ bias,
             float* __restrict__ C, int ldc,
             float* __restrict__ C2, int ldc2,
             int M, int N, int K) {
  const int lane  = threadIdx.x;
  const int tileN = blockIdx.x * 32;
  const int tileM = blockIdx.y * 32;
  const bool hasN1 = (tileN + 16) < N;  // half-tile predicate (N=80 case)

  v8f acc00 = {}; v8f acc01 = {}; v8f acc10 = {}; v8f acc11 = {};
  const int kfull = K & ~31;
  for (int k0 = 0; k0 < kfull; k0 += 32) {
    __builtin_prefetch(A + (size_t)(tileM + (lane & 15)) * lda + k0 + 64, 0, 1);
    v16bf a0 = frag_a_(A, lda, tileM,      k0, K, lane, false);
    v16bf a1 = frag_a_(A, lda, tileM + 16, k0, K, lane, false);
    v16bf b0 = frag_b_(W, ldw, tileN,      k0, K, lane, false);
    acc00 = wmma_bf16_(a0, b0, acc00);
    acc10 = wmma_bf16_(a1, b0, acc10);
    if (hasN1) {
      v16bf b1 = frag_b_(W, ldw, tileN + 16, k0, K, lane, false);
      acc01 = wmma_bf16_(a0, b1, acc01);
      acc11 = wmma_bf16_(a1, b1, acc11);
    }
  }
  if (kfull < K) {  // K tail (DT_RANK = 48), zero-padded
    int k0 = kfull;
    v16bf a0 = frag_a_(A, lda, tileM,      k0, K, lane, true);
    v16bf a1 = frag_a_(A, lda, tileM + 16, k0, K, lane, true);
    v16bf b0 = frag_b_(W, ldw, tileN,      k0, K, lane, true);
    acc00 = wmma_bf16_(a0, b0, acc00);
    acc10 = wmma_bf16_(a1, b0, acc10);
    if (hasN1) {
      v16bf b1 = frag_b_(W, ldw, tileN + 16, k0, K, lane, true);
      acc01 = wmma_bf16_(a0, b1, acc01);
      acc11 = wmma_bf16_(a1, b1, acc11);
    }
  }
  store_tile_<EPI>(acc00, C, ldc, C2, ldc2, bias, tileM,      tileN,      lane);
  store_tile_<EPI>(acc10, C, ldc, C2, ldc2, bias, tileM + 16, tileN,      lane);
  if (hasN1) {
    store_tile_<EPI>(acc01, C, ldc, C2, ldc2, bias, tileM,      tileN + 16, lane);
    store_tile_<EPI>(acc11, C, ldc, C2, ldc2, bias, tileM + 16, tileN + 16, lane);
  }
}

// ---------------------------------------------------------------------------
// TDM-staged WMMA GEMM: 4 waves / 64x64 macro-tile. Wave 0 issues two 2-D
// tensor_load_to_lds (A tile 64x32, W tile 64x32), waits TENSORcnt, then all
// waves build bf16 fragments from LDS and run 4 WMMAs each per K-step.
// Requires M%64==0, N%64==0, K%32==0.
// ---------------------------------------------------------------------------
template <int EPI>
__global__ void __launch_bounds__(128)
gemm_wmma_tdm(const float* __restrict__ A, int lda,
              const float* __restrict__ W, int ldw,
              float* __restrict__ C, int ldc,
              float* __restrict__ C2, int ldc2,
              int M, int N, int K) {
  __shared__ float sA[64 * 32];
  __shared__ float sW[64 * 32];
  const int lane = threadIdx.x & 31;
  const int wave = threadIdx.x >> 5;      // 0..3
  const int wm = (wave >> 1) * 32;        // wave's M offset within macro-tile
  const int wn = (wave & 1) * 32;         // wave's N offset within macro-tile
  const int blockM = blockIdx.y * 64;
  const int blockN = blockIdx.x * 64;
  const unsigned ldsA = (unsigned)(uintptr_t)(void*)sA;
  const unsigned ldsW = (unsigned)(uintptr_t)(void*)sW;

  v8f acc00 = {}; v8f acc01 = {}; v8f acc10 = {}; v8f acc11 = {};
  for (int k0 = 0; k0 < K; k0 += 32) {
    if (wave == 0) {
      tdm_load_2d_(ldsA, A + (size_t)blockM * lda + k0, 64, 32, lda);
      tdm_load_2d_(ldsW, W + (size_t)blockN * ldw + k0, 64, 32, ldw);
      __builtin_amdgcn_s_wait_tensorcnt(0);
    }
    __syncthreads();  // LDS tiles visible to all waves

    v16bf a0 = frag_a_(sA, 32, wm,      0, 32, lane, false);
    v16bf a1 = frag_a_(sA, 32, wm + 16, 0, 32, lane, false);
    v16bf b0 = frag_b_(sW, 32, wn,      0, 32, lane, false);
    v16bf b1 = frag_b_(sW, 32, wn + 16, 0, 32, lane, false);
    acc00 = wmma_bf16_(a0, b0, acc00);
    acc10 = wmma_bf16_(a1, b0, acc10);
    acc01 = wmma_bf16_(a0, b1, acc01);
    acc11 = wmma_bf16_(a1, b1, acc11);

    __syncthreads();  // protect LDS against next TDM overwrite
  }
  store_tile_<EPI>(acc00, C, ldc, C2, ldc2, nullptr, blockM + wm,      blockN + wn,      lane);
  store_tile_<EPI>(acc10, C, ldc, C2, ldc2, nullptr, blockM + wm + 16, blockN + wn,      lane);
  store_tile_<EPI>(acc01, C, ldc, C2, ldc2, nullptr, blockM + wm,      blockN + wn + 16, lane);
  store_tile_<EPI>(acc11, C, ldc, C2, ldc2, nullptr, blockM + wm + 16, blockN + wn + 16, lane);
}

__global__ void embed_kernel(const int* __restrict__ ids,
                             const float* __restrict__ embed,
                             float* __restrict__ x, int flip) {
  int i = blockIdx.x * blockDim.x + threadIdx.x;
  if (i >= NTOK * D_MODEL) return;
  int row = i / D_MODEL, j = i - row * D_MODEL;
  int b = row / SEQ, l = row - b * SEQ;
  int sl = flip ? (SEQ - 1 - l) : l;
  int id = ids[b * SEQ + sl];
  x[i] = embed[(size_t)id * D_MODEL + j];
}

__global__ void rmsnorm_kernel(const float* __restrict__ x,
                               const float* __restrict__ w,
                               float* __restrict__ h) {
  __shared__ float red[256];
  int row = blockIdx.x;
  const float* xr = x + (size_t)row * D_MODEL;
  float ss = 0.f;
  for (int i = threadIdx.x; i < D_MODEL; i += 256) { float v = xr[i]; ss += v * v; }
  red[threadIdx.x] = ss;
  __syncthreads();
  for (int s = 128; s > 0; s >>= 1) {
    if (threadIdx.x < s) red[threadIdx.x] += red[threadIdx.x + s];
    __syncthreads();
  }
  float scale = rsqrtf(red[0] / (float)D_MODEL + 1e-5f);
  for (int i = threadIdx.x; i < D_MODEL; i += 256)
    h[(size_t)row * D_MODEL + i] = xr[i] * scale * w[i];
}

__global__ void conv_silu_kernel(const float* __restrict__ xz,
                                 const float* __restrict__ cw,
                                 const float* __restrict__ cb,
                                 float* __restrict__ xc) {
  int i = blockIdx.x * blockDim.x + threadIdx.x;
  if (i >= NTOK * D_INNER) return;
  int row = i / D_INNER, c = i - row * D_INNER;
  int b = row / SEQ, l = row - b * SEQ;
  float acc = cb[c];
  const float* wk = cw + (size_t)c * D_CONV;
#pragma unroll
  for (int k = 0; k < D_CONV; ++k) {
    int ls = l - (D_CONV - 1) + k;
    if (ls >= 0)
      acc += xz[((size_t)(b * SEQ + ls)) * (2 * D_INNER) + c] * wk[k];
  }
  xc[i] = silu_(acc);
}

// One thread per (batch, d_inner) channel; 16-state recurrence in registers.
// Fuses y = (scan + x*D) * silu(z).
__global__ void scan_kernel(const float* __restrict__ delta,
                            const float* __restrict__ xc,
                            const float* __restrict__ dbl,
                            const float* __restrict__ xz,
                            const float* __restrict__ A_log,
                            const float* __restrict__ Dp,
                            float* __restrict__ y) {
  int t = blockIdx.x * blockDim.x + threadIdx.x;
  if (t >= BATCH * D_INNER) return;
  int b = t / D_INNER, d = t - b * D_INNER;
  float Ar[D_STATE];
#pragma unroll
  for (int n = 0; n < D_STATE; ++n) Ar[n] = -__expf(A_log[(size_t)d * D_STATE + n]);
  float Dv = Dp[d];
  float s[D_STATE];
#pragma unroll
  for (int n = 0; n < D_STATE; ++n) s[n] = 0.f;
  for (int l = 0; l < SEQ; ++l) {
    size_t row = (size_t)b * SEQ + l;
    float dt = delta[row * D_INNER + d];
    float xv = xc[row * D_INNER + d];
    float zv = xz[row * (2 * D_INNER) + D_INNER + d];
    const float* Br = dbl + row * XPROJ_N + DT_RANK;
    const float* Cr = Br + D_STATE;
    float dx = dt * xv;
    float acc = 0.f;
#pragma unroll
    for (int n = 0; n < D_STATE; ++n) {
      s[n] = s[n] * __expf(dt * Ar[n]) + dx * Br[n];
      acc += s[n] * Cr[n];
    }
    y[row * D_INNER + d] = (acc + xv * Dv) * silu_(zv);
  }
}

extern "C" void kernel_launch(void* const* d_in, const int* in_sizes, int n_in,
                              void* d_out, int out_size, void* d_ws, size_t ws_size,
                              hipStream_t stream) {
  (void)in_sizes; (void)n_in; (void)out_size; (void)ws_size;
  const int* ids = (const int*)d_in[0];

  float* ws = (float*)d_ws;
  size_t o = 0;
  float* x     = ws + o; o += (size_t)NTOK * D_MODEL;
  float* h     = ws + o; o += (size_t)NTOK * D_MODEL;
  float* xz    = ws + o; o += (size_t)NTOK * 2 * D_INNER;
  float* xc    = ws + o; o += (size_t)NTOK * D_INNER;
  float* dbl   = ws + o; o += (size_t)NTOK * XPROJ_N;
  float* delta = ws + o; o += (size_t)NTOK * D_INNER;
  float* y     = ws + o; o += (size_t)NTOK * D_INNER;

  for (int dir = 0; dir < 2; ++dir) {
    const float* p_embed    = (const float*)d_in[1 + dir * 11 + 0];
    const float* p_norm_w   = (const float*)d_in[1 + dir * 11 + 1];
    const float* p_in_proj  = (const float*)d_in[1 + dir * 11 + 2];
    const float* p_conv_w   = (const float*)d_in[1 + dir * 11 + 3];
    const float* p_conv_b   = (const float*)d_in[1 + dir * 11 + 4];
    const float* p_x_proj   = (const float*)d_in[1 + dir * 11 + 5];
    const float* p_dt_w     = (const float*)d_in[1 + dir * 11 + 6];
    const float* p_dt_b     = (const float*)d_in[1 + dir * 11 + 7];
    const float* p_A_log    = (const float*)d_in[1 + dir * 11 + 8];
    const float* p_Dp       = (const float*)d_in[1 + dir * 11 + 9];
    const float* p_out_proj = (const float*)d_in[1 + dir * 11 + 10];

    embed_kernel<<<(NTOK * D_MODEL + 255) / 256, 256, 0, stream>>>(ids, p_embed, x, dir);

    for (int layer = 0; layer < N_LAYER; ++layer) {
      rmsnorm_kernel<<<NTOK, 256, 0, stream>>>(x, p_norm_w + (size_t)layer * D_MODEL, h);

      // xz[2048,3072] = h[2048,768] * in_proj[3072,768]^T  (TDM-staged path)
      gemm_wmma_tdm<0><<<dim3(2 * D_INNER / 64, NTOK / 64), 128, 0, stream>>>(
          h, D_MODEL, p_in_proj + (size_t)layer * 2 * D_INNER * D_MODEL, D_MODEL,
          xz, 2 * D_INNER, nullptr, 0, NTOK, 2 * D_INNER, D_MODEL);

      conv_silu_kernel<<<(NTOK * D_INNER + 255) / 256, 256, 0, stream>>>(
          xz, p_conv_w + (size_t)layer * D_INNER * D_CONV,
          p_conv_b + (size_t)layer * D_INNER, xc);

      // dbl[2048,80] = xc[2048,1536] * x_proj[80,1536]^T  (ragged N -> direct)
      gemm_wmma_nt<0><<<dim3((XPROJ_N + 31) / 32, NTOK / 32), 32, 0, stream>>>(
          xc, D_INNER, p_x_proj + (size_t)layer * XPROJ_N * D_INNER, D_INNER,
          nullptr, dbl, XPROJ_N, nullptr, 0, NTOK, XPROJ_N, D_INNER);

      // delta[2048,1536] = softplus(dbl[:, :48] * dt_w[1536,48]^T + dt_b)
      gemm_wmma_nt<1><<<dim3(D_INNER / 32, NTOK / 32), 32, 0, stream>>>(
          dbl, XPROJ_N, p_dt_w + (size_t)layer * D_INNER * DT_RANK, DT_RANK,
          p_dt_b + (size_t)layer * D_INNER, delta, D_INNER, nullptr, 0,
          NTOK, D_INNER, DT_RANK);

      scan_kernel<<<(BATCH * D_INNER + 255) / 256, 256, 0, stream>>>(
          delta, xc, dbl, xz, p_A_log + (size_t)layer * D_INNER * D_STATE,
          p_Dp + (size_t)layer * D_INNER, y);

      // x_next[2048,768] = y * out_proj[768,1536]^T ; dual-store into d_out
      float* out_slice = (float*)d_out +
          (size_t)layer * BATCH * SEQ * (2 * D_MODEL) + (size_t)dir * D_MODEL;
      gemm_wmma_tdm<2><<<dim3(D_MODEL / 64, NTOK / 64), 128, 0, stream>>>(
          y, D_INNER, p_out_proj + (size_t)layer * D_MODEL * D_INNER, D_INNER,
          x, D_MODEL, out_slice, 2 * D_MODEL, NTOK, D_MODEL, D_INNER);
    }
  }
}
